// ModelRNN_39419209842803
// MI455X (gfx1250) — matrix operand order repaired and checked by reading, not compiled, not run on hardware
//
#include <hip/hip_runtime.h>
#include <math.h>

// ---------------------------------------------------------------------------
// Problem dimensions (compile-time constants, from the reference)
// ---------------------------------------------------------------------------
#define BATCH 64
#define SEQ   1024
#define CH    512      // cnn feature dim
#define VOC   140
#define ATT   512      // attn size
#define HID   512      // hidden size
#define TSTEP 128
#define GATES4 (4*HID)         // 2048
#define XPAD  672              // V + A = 652 padded to multiple of 32

typedef __bf16 bhalf;
typedef __attribute__((ext_vector_type(16))) __bf16 v16bf;
typedef __attribute__((ext_vector_type(8)))  float   v8f;

#define DEVINL __device__ __forceinline__

// ---------------------------------------------------------------------------
// bf16 <-> f32 helpers (bit-level, no dependence on __bf16 arithmetic support)
// ---------------------------------------------------------------------------
DEVINL bhalf f2bf(float f) {
    union { float f; unsigned u; } c; c.f = f;
    union { unsigned short s; bhalf b; } r; r.s = (unsigned short)(c.u >> 16);
    return r.b;
}
DEVINL float bf2f(bhalf b) {
    union { bhalf b; unsigned short s; } c; c.b = b;
    union { unsigned u; float f; } r; r.u = ((unsigned)c.s) << 16;
    return r.f;
}
DEVINL float bfu_lo(unsigned pk) {
    union { unsigned u; float f; } r; r.u = pk << 16; return r.f;
}
DEVINL float bfu_hi(unsigned pk) {
    union { unsigned u; float f; } r; r.u = pk & 0xFFFF0000u; return r.f;
}
DEVINL float sigm(float x) { return 1.f / (1.f + __expf(-x)); }

// ---------------------------------------------------------------------------
// WMMA fragment loads per CDNA5 ISA 16-bit layouts (CDNA5 ISA 7.12.2)
// A (16x32, MxK): lane L holds row M=L&15; K in {8*(L>>4)+0..7, 16+8*(L>>4)+0..7}
// B (32x16, KxN): lane L holds col N=L&15; K in {8*(L>>4)+0..7, 16+8*(L>>4)+0..7}
// C/D (16x16 f32): lane L holds col N=L&15; VGPR g holds row M = g + 8*(L>>4)
// ---------------------------------------------------------------------------
DEVINL v16bf load_a_bf16(const bhalf* A, int lda, int row0, int k0, int lane) {
    const bhalf* p = A + (size_t)(row0 + (lane & 15)) * lda + k0 + ((lane >> 4) << 3);
    v16bf r;
#pragma unroll
    for (int e = 0; e < 8; ++e) { r[e] = p[e]; r[e + 8] = p[e + 16]; }
    return r;
}
DEVINL v16bf load_a_f32cvt(const float* A, int lda, int row0, int k0, int lane) {
    const float* p = A + (size_t)(row0 + (lane & 15)) * lda + k0 + ((lane >> 4) << 3);
    v16bf r;
#pragma unroll
    for (int e = 0; e < 8; ++e) { r[e] = f2bf(p[e]); r[e + 8] = f2bf(p[e + 16]); }
    return r;
}
// W stored [N_total, K_total] row-major; logical B[k][n] = W[n][k]
DEVINL v16bf load_b_bf16(const bhalf* W, int ldw, int col0, int k0, int lane) {
    const bhalf* p = W + (size_t)(col0 + (lane & 15)) * ldw + k0 + ((lane >> 4) << 3);
    v16bf r;
#pragma unroll
    for (int e = 0; e < 8; ++e) { r[e] = p[e]; r[e + 8] = p[e + 16]; }
    return r;
}

// ---------------------------------------------------------------------------
// Setup kernels: fp32 -> bf16 weight conversion (+zero-pad for W_ih), init
// ---------------------------------------------------------------------------
__global__ void cvt_kernel(const float* __restrict__ src, bhalf* __restrict__ dst, int n) {
    int i = blockIdx.x * blockDim.x + threadIdx.x;
    if (i < n) dst[i] = f2bf(src[i]);
}

__global__ void cvt_pad_kernel(const float* __restrict__ src, bhalf* __restrict__ dst,
                               int rows, int sc, int dc) {
    int i = blockIdx.x * blockDim.x + threadIdx.x;
    if (i < rows * dc) {
        int r = i / dc, cidx = i - r * dc;
        dst[i] = f2bf(cidx < sc ? src[(size_t)r * sc + cidx] : 0.f);
    }
}

__global__ void init_kernel(const int* __restrict__ labels, bhalf* __restrict__ xB,
                            float* __restrict__ hS, float* __restrict__ cS,
                            bhalf* __restrict__ hB) {
    int i = blockIdx.x * blockDim.x + threadIdx.x;
    if (i < BATCH * XPAD) {
        int b = i / XPAD, cidx = i - b * XPAD;
        float v = 0.f;
        if (cidx < VOC && cidx == labels[(size_t)b * TSTEP]) v = 1.f;  // <sos> one-hot
        xB[i] = f2bf(v);
    }
    if (i < BATCH * HID) {
        hS[i] = 0.f; cS[i] = 0.f; hB[i] = f2bf(0.f);
    }
}

// ---------------------------------------------------------------------------
// keys[m, a] = sum_c img[m,c] * Icw[a,c] + Ic_b[a]   (65536 x 512, K=512)
// A converted fp32->bf16 in-register; result stored bf16 (64 MB, L2-resident)
// 8 waves/block, one 16x16 tile per wave.
// ---------------------------------------------------------------------------
__global__ void keys_gemm_kernel(const float* __restrict__ img, const bhalf* __restrict__ IcwB,
                                 const float* __restrict__ Icb, bhalf* __restrict__ keys) {
    const int lane = threadIdx.x & 31;
    const int tile = blockIdx.x * 8 + (threadIdx.x >> 5);
    const int mt = tile >> 5;          // 4096 row tiles
    const int nt = tile & 31;          // 32 col tiles
    const int row0 = mt * 16, col0 = nt * 16;
    v8f acc = {};
    for (int k0 = 0; k0 < CH; k0 += 32) {
        v16bf a = load_a_f32cvt(img, CH, row0, k0, lane);
        v16bf b = load_b_bf16(IcwB, CH, col0, k0, lane);
        acc = __builtin_amdgcn_wmma_f32_16x16x32_bf16(false, a, false, b, (short)0, acc, false, false);
    }
    const int n = col0 + (lane & 15);
    const float bv = Icb[n];
    const int mbase = row0 + ((lane >> 4) << 3);
#pragma unroll
    for (int g = 0; g < 8; ++g)
        keys[(size_t)(mbase + g) * ATT + n] = f2bf(acc[g] + bv);
}

// ---------------------------------------------------------------------------
// q[b, a] = sum_j h[b,j] * Hcw[a,j] + Hc_b[a]    (64 x 512, K=512)
// ---------------------------------------------------------------------------
__global__ void q_gemm_kernel(const bhalf* __restrict__ hB, const bhalf* __restrict__ HcwB,
                              const float* __restrict__ Hcb, float* __restrict__ q) {
    const int lane = threadIdx.x & 31;
    const int tile = blockIdx.x * 8 + (threadIdx.x >> 5);   // 128 tiles
    const int mt = tile >> 5;          // 4 row tiles
    const int nt = tile & 31;          // 32 col tiles
    const int row0 = mt * 16, col0 = nt * 16;
    v8f acc = {};
    for (int k0 = 0; k0 < HID; k0 += 32) {
        v16bf a = load_a_bf16(hB, HID, row0, k0, lane);
        v16bf b = load_b_bf16(HcwB, HID, col0, k0, lane);
        acc = __builtin_amdgcn_wmma_f32_16x16x32_bf16(false, a, false, b, (short)0, acc, false, false);
    }
    const int n = col0 + (lane & 15);
    const float bv = Hcb[n];
    const int mbase = row0 + ((lane >> 4) << 3);
#pragma unroll
    for (int g = 0; g < 8; ++g)
        q[(size_t)(mbase + g) * ATT + n] = acc[g] + bv;
}

// ---------------------------------------------------------------------------
// Attention: scores = q . keys / sqrt(A); softmax over S; ctx = w . keys
// One block per batch element; ctx written as bf16 into x[b, VOC..VOC+ATT)
// ---------------------------------------------------------------------------
__global__ void attn_kernel(const float* __restrict__ q, const bhalf* __restrict__ keys,
                            bhalf* __restrict__ xB) {
    const int b = blockIdx.x;
    const int tid = threadIdx.x;
    __shared__ float q_s[ATT];
    __shared__ float sc[SEQ];
    __shared__ float red[256];

    for (int i = tid; i < ATT; i += 256) q_s[i] = q[(size_t)b * ATT + i];
    __syncthreads();

    float sc4[4];
    float lmax = -3.402823466e38f;
#pragma unroll
    for (int j = 0; j < 4; ++j) {
        const int s = tid + 256 * j;
        const bhalf* kr = keys + ((size_t)b * SEQ + s) * ATT;
        float acc = 0.f;
        for (int a = 0; a < ATT; ++a) acc = fmaf(bf2f(kr[a]), q_s[a], acc);
        acc *= 0.044194173824159216f;   // 1/sqrt(512)
        sc4[j] = acc;
        lmax = fmaxf(lmax, acc);
    }
    red[tid] = lmax; __syncthreads();
    for (int st = 128; st > 0; st >>= 1) {
        if (tid < st) red[tid] = fmaxf(red[tid], red[tid + st]);
        __syncthreads();
    }
    const float mx = red[0];
    __syncthreads();
    float lsum = 0.f;
#pragma unroll
    for (int j = 0; j < 4; ++j) {
        const float e = __expf(sc4[j] - mx);
        sc[tid + 256 * j] = e;
        lsum += e;
    }
    red[tid] = lsum; __syncthreads();
    for (int st = 128; st > 0; st >>= 1) {
        if (tid < st) red[tid] += red[tid + st];
        __syncthreads();
    }
    const float inv = 1.f / red[0];

    // context: thread owns attention columns (2*tid, 2*tid+1); dword loads
    const int a0 = tid * 2;
    float acc0 = 0.f, acc1 = 0.f;
    for (int s = 0; s < SEQ; ++s) {
        const float w = sc[s] * inv;
        const unsigned* kp = (const unsigned*)(keys + ((size_t)b * SEQ + s) * ATT);
        const unsigned pk = kp[tid];
        acc0 = fmaf(w, bfu_lo(pk), acc0);
        acc1 = fmaf(w, bfu_hi(pk), acc1);
    }
    xB[(size_t)b * XPAD + VOC + a0]     = f2bf(acc0);
    xB[(size_t)b * XPAD + VOC + a0 + 1] = f2bf(acc1);
}

// ---------------------------------------------------------------------------
// gates[b, n] = sum_k x[b,k] Wih[n,k] + sum_j h[b,j] Whh[n,j] + b_ih + b_hh
// (64 x 2048, K = 672(pad) + 512)
// ---------------------------------------------------------------------------
__global__ void gates_gemm_kernel(const bhalf* __restrict__ xB, const bhalf* __restrict__ WihB,
                                  const bhalf* __restrict__ hB, const bhalf* __restrict__ WhhB,
                                  const float* __restrict__ bih, const float* __restrict__ bhh,
                                  float* __restrict__ gates) {
    const int lane = threadIdx.x & 31;
    const int tile = blockIdx.x * 8 + (threadIdx.x >> 5);   // 512 tiles
    const int mt = tile >> 7;          // 4 row tiles
    const int nt = tile & 127;         // 128 col tiles
    const int row0 = mt * 16, col0 = nt * 16;
    v8f acc = {};
    for (int k0 = 0; k0 < XPAD; k0 += 32) {
        v16bf a = load_a_bf16(xB, XPAD, row0, k0, lane);
        v16bf b = load_b_bf16(WihB, XPAD, col0, k0, lane);
        acc = __builtin_amdgcn_wmma_f32_16x16x32_bf16(false, a, false, b, (short)0, acc, false, false);
    }
    for (int k0 = 0; k0 < HID; k0 += 32) {
        v16bf a = load_a_bf16(hB, HID, row0, k0, lane);
        v16bf b = load_b_bf16(WhhB, HID, col0, k0, lane);
        acc = __builtin_amdgcn_wmma_f32_16x16x32_bf16(false, a, false, b, (short)0, acc, false, false);
    }
    const int n = col0 + (lane & 15);
    const float bv = bih[n] + bhh[n];
    const int mbase = row0 + ((lane >> 4) << 3);
#pragma unroll
    for (int g = 0; g < 8; ++g)
        gates[(size_t)(mbase + g) * GATES4 + n] = acc[g] + bv;
}

// ---------------------------------------------------------------------------
// LSTM pointwise + output projection + argmax one-hot. One block per batch.
// ---------------------------------------------------------------------------
__global__ void cell_kernel(const float* __restrict__ gates, float* __restrict__ cS,
                            float* __restrict__ hS, bhalf* __restrict__ hB,
                            const float* __restrict__ Wout, const float* __restrict__ bout,
                            bhalf* __restrict__ xB, float* __restrict__ out, int t) {
    const int b = blockIdx.x;
    const int tid = threadIdx.x;
    __shared__ float h_s[HID];
    __shared__ float lo[VOC];
    __shared__ int amax_s;

    for (int n = tid; n < HID; n += 256) {
        const size_t gb = (size_t)b * GATES4;
        const float gi = gates[gb + n];
        const float gf = gates[gb + HID + n];
        const float gg = gates[gb + 2 * HID + n];
        const float go = gates[gb + 3 * HID + n];
        const size_t hi = (size_t)b * HID + n;
        const float cn = sigm(gf) * cS[hi] + sigm(gi) * tanhf(gg);
        const float hn = sigm(go) * tanhf(cn);
        cS[hi] = cn;
        hS[hi] = hn;
        hB[hi] = f2bf(hn);
        h_s[n] = hn;
    }
    __syncthreads();

    if (tid < VOC) {
        const float* wr = Wout + (size_t)tid * HID;
        float acc = bout[tid];
        for (int a = 0; a < HID; ++a) acc = fmaf(wr[a], h_s[a], acc);
        lo[tid] = acc;
        out[((size_t)b * TSTEP + t) * VOC + tid] = acc;
    }
    __syncthreads();
    if (tid == 0) {
        int best = 0;
        for (int v = 1; v < VOC; ++v) if (lo[v] > lo[best]) best = v;   // first-max, like jnp.argmax
        amax_s = best;
    }
    __syncthreads();
    if (tid < VOC)
        xB[(size_t)b * XPAD + tid] = f2bf(tid == amax_s ? 1.f : 0.f);
}

// ---------------------------------------------------------------------------
// Host launcher
// ---------------------------------------------------------------------------
extern "C" void kernel_launch(void* const* d_in, const int* in_sizes, int n_in,
                              void* d_out, int out_size, void* d_ws, size_t ws_size,
                              hipStream_t stream) {
    (void)in_sizes; (void)n_in; (void)out_size; (void)ws_size;

    const float* img    = (const float*)d_in[0];   // [B,S,C]
    const int*   labels = (const int*)  d_in[1];   // [B,T]
    const float* Icw    = (const float*)d_in[2];   // [A,C]
    const float* Icb    = (const float*)d_in[3];   // [A]
    const float* Hcw    = (const float*)d_in[4];   // [A,H]
    const float* Hcb    = (const float*)d_in[5];   // [A]
    const float* Wih    = (const float*)d_in[6];   // [4H, V+A]
    const float* bih    = (const float*)d_in[7];   // [4H]
    const float* Whh    = (const float*)d_in[8];   // [4H, H]
    const float* bhh    = (const float*)d_in[9];   // [4H]
    const float* Wout   = (const float*)d_in[10];  // [V,H]
    const float* bout   = (const float*)d_in[11];  // [V]
    float* out = (float*)d_out;                    // [B,T,V]

    char* ws = (char*)d_ws;
    size_t off = 0;
    auto wsalloc = [&](size_t bytes) -> void* {
        void* p = ws + off;
        off += (bytes + 255) & ~(size_t)255;
        return p;
    };
    bhalf* keysB = (bhalf*)wsalloc(2ull * BATCH * SEQ * ATT);   // 64 MB (L2-resident)
    bhalf* IcwB  = (bhalf*)wsalloc(2ull * ATT * CH);
    bhalf* HcwB  = (bhalf*)wsalloc(2ull * ATT * HID);
    bhalf* WihB  = (bhalf*)wsalloc(2ull * GATES4 * XPAD);       // zero-padded cols
    bhalf* WhhB  = (bhalf*)wsalloc(2ull * GATES4 * HID);
    float* qbuf  = (float*)wsalloc(4ull * BATCH * ATT);
    float* gates = (float*)wsalloc(4ull * BATCH * GATES4);
    float* hS    = (float*)wsalloc(4ull * BATCH * HID);
    float* cS    = (float*)wsalloc(4ull * BATCH * HID);
    bhalf* hB    = (bhalf*)wsalloc(2ull * BATCH * HID);
    bhalf* xB    = (bhalf*)wsalloc(2ull * BATCH * XPAD);

    // --- one-time setup ---
    cvt_kernel<<<(ATT * CH + 255) / 256, 256, 0, stream>>>(Icw, IcwB, ATT * CH);
    cvt_kernel<<<(ATT * HID + 255) / 256, 256, 0, stream>>>(Hcw, HcwB, ATT * HID);
    cvt_kernel<<<(GATES4 * HID + 255) / 256, 256, 0, stream>>>(Whh, WhhB, GATES4 * HID);
    cvt_pad_kernel<<<(GATES4 * XPAD + 255) / 256, 256, 0, stream>>>(Wih, WihB, GATES4, VOC + ATT, XPAD);
    init_kernel<<<(BATCH * XPAD + 255) / 256, 256, 0, stream>>>(labels, xB, hS, cS, hB);

    // keys projection: 4096 x 32 tiles, 8 waves/block
    keys_gemm_kernel<<<(4096 * 32) / 8, 256, 0, stream>>>(img, IcwB, Icb, keysB);

    // --- sequential decode loop ---
    for (int t = 0; t < TSTEP; ++t) {
        q_gemm_kernel<<<(4 * 32) / 8, 256, 0, stream>>>(hB, HcwB, Hcb, qbuf);
        attn_kernel<<<BATCH, 256, 0, stream>>>(qbuf, keysB, xB);
        gates_gemm_kernel<<<(4 * 128) / 8, 256, 0, stream>>>(xB, WihB, hB, WhhB, bih, bhh, gates);
        cell_kernel<<<BATCH, 256, 0, stream>>>(gates, cS, hS, hB, Wout, bout, xB, out, t);
    }
}
